// GraphAttention_7653631721779
// MI455X (gfx1250) — matrix operand order repaired and checked by reading, not compiled, not run on hardware
//
#include <hip/hip_runtime.h>
#include <hip/hip_bf16.h>

// ---------------------------------------------------------------------------
// GraphNet layer for MI455X (gfx1250), wave32, WMMA f32_16x16x32_f16 path.
//
// v2: 64-row M-blocking everywhere — each B (weight) fragment load from L2
// feeds 4 consecutive WMMAs (4 M-tiles per wave), quadrupling arithmetic
// intensity on the weight stream. Branchless epilogues via template.
// ---------------------------------------------------------------------------

typedef __attribute__((ext_vector_type(16))) _Float16 v16h;
typedef __attribute__((ext_vector_type(8)))  _Float16 v8h;
typedef __attribute__((ext_vector_type(8)))  float    v8f;

#define F 128   // FD == FS == 128

__device__ __forceinline__ v8f wmma16(v16h a, v16h b, v8f c) {
  return __builtin_amdgcn_wmma_f32_16x16x32_f16(false, a, false, b,
                                                (short)0, c, false, false);
}

// ---- helper: load a 16x32 f16 A-fragment for this lane from an LDS tile ----
// A layout (ISA 05_wmma): lane L holds row M = L&15.
//   L<16 : K = kb*32 + {0..7, 16..23}
//   L>=16: K = kb*32 + {8..15, 24..31}
__device__ __forceinline__ v16h load_a_frag(const _Float16* tile, int row_stride,
                                            int kb, int lane) {
  int m  = lane & 15;
  int kh = (lane < 16) ? 0 : 8;
  const _Float16* base = tile + m * row_stride + kb * 32 + kh;
  union { v16h v; v8h h[2]; } u;
  u.h[0] = *(const v8h*)(base);        // K = kb*32 + kh + [0..7]
  u.h[1] = *(const v8h*)(base + 16);   // K = kb*32 + kh + [16..23]
  return u.v;
}

// ---- pack fp32 weight [K,N] (row-major) into WMMA-B lane layout (f16) ------
// Packed element tid = ((nt*KB + kb)*32 + lane)*16 + i maps to
//   k = kb*32 + (lane>=16 ? 8 : 0) + (i&7) + (i>=8 ? 16 : 0),  n = nt*16 + (lane&15)
// so lane L of a wave reading tile (nt,kb) fetches 32 contiguous bytes.
__global__ void pack_b(const float* __restrict__ W, _Float16* __restrict__ P,
                       int K, int N) {
  int tid = blockIdx.x * blockDim.x + threadIdx.x;
  if (tid >= K * N) return;
  int KB   = K >> 5;
  int i    = tid & 15;
  int lane = (tid >> 4) & 31;
  int g    = tid >> 9;
  int kb   = g % KB;
  int nt   = g / KB;
  int k = kb * 32 + ((lane >> 4) << 3) + (i & 7) + ((i >> 3) << 4);
  int n = nt * 16 + (lane & 15);
  P[tid] = (_Float16)W[k * N + n];
}

// ---- node encoder: enc = relu(X @ W + b); 64 rows/block, 4 M-tiles/wave ----
template <bool WRITE_F32>
__global__ __launch_bounds__(256) void node_encode(
    const float* __restrict__ X,          // [rows, 128] fp32
    const _Float16* __restrict__ Wp,      // packed [128,128]
    const float* __restrict__ bias,       // [128]
    _Float16* __restrict__ out_f16,       // [rows, 128]
    float* __restrict__ out_f32) {        // dst path: aggregator init
  __shared__ __align__(16) _Float16 A[64 * F];   // 16 KB
  int row0 = blockIdx.x * 64;
  int tid  = threadIdx.x;
  // stage & convert 64x128 fp32 -> f16 in LDS
#pragma unroll
  for (int it = 0; it < 4; ++it) {
    int r  = it * 16 + (tid >> 4);
    int c0 = (tid & 15) * 8;
    const float* src = X + (size_t)(row0 + r) * F + c0;
#pragma unroll
    for (int j = 0; j < 8; ++j) A[r * F + c0 + j] = (_Float16)src[j];
  }
  __syncthreads();
  int wave = tid >> 5, lane = tid & 31;
  int mlow = lane & 15;
  int mh   = (lane < 16) ? 0 : 8;     // C/D: M = r + mh
  v8f acc[4] = {};
#pragma unroll
  for (int kb = 0; kb < 4; ++kb) {
    v16h b = *(const v16h*)(Wp + ((size_t)(wave * 4 + kb) * 32 + lane) * 16);
#pragma unroll
    for (int mt = 0; mt < 4; ++mt)
      acc[mt] = wmma16(load_a_frag(A + mt * 16 * F, F, kb, lane), b, acc[mt]);
  }
  int col = wave * 16 + mlow;
  float bv = bias[col];
#pragma unroll
  for (int mt = 0; mt < 4; ++mt) {
#pragma unroll
    for (int r = 0; r < 8; ++r) {
      float v = acc[mt][r] + bv;
      v = v > 0.f ? v : 0.f;
      size_t o = (size_t)(row0 + mt * 16 + r + mh) * F + col;
      out_f16[o] = (_Float16)v;
      if (WRITE_F32) out_f32[o] = v;
    }
  }
}

// ---- edge MLP: 64 edges/block; each B load feeds 4 WMMAs -------------------
__global__ __launch_bounds__(256) void edge_mlp(
    const _Float16* __restrict__ src_enc, const _Float16* __restrict__ dst_enc,
    const float* __restrict__ src_pos,    const float* __restrict__ dst_pos,
    const float* __restrict__ dist_w,     const float* __restrict__ dist_b,
    const _Float16* __restrict__ W1p,     const float* __restrict__ b1,
    const _Float16* __restrict__ W2p,     const float* __restrict__ b2,
    const int* __restrict__ esrc,         const int* __restrict__ edst,
    float* __restrict__ agg, int E) {
  __shared__ __align__(16) _Float16 A[64 * 384];   // 48 KB, [64 edges, 384]
  __shared__ __align__(16) _Float16 H[64 * F];     // 16 KB, hidden relu tile
  __shared__ int sidx[64], didx[64];
  int e0  = blockIdx.x * 64;
  int tid = threadIdx.x;
  if (tid < 64) {
    int e = e0 + tid;
    sidx[tid] = (e < E) ? esrc[e] : -1;
    didx[tid] = (e < E) ? edst[e] : -1;
  }
  __syncthreads();
  // gather src/dst rows (16B chunks) + compute dist features
#pragma unroll
  for (int it = 0; it < 4; ++it) {
    int e  = it * 16 + (tid >> 4);
    int c  = (tid & 15) * 8;            // 8 f16 = 16 bytes
    int si = sidx[e], di = didx[e];
    uint4 z = {0, 0, 0, 0};
    uint4 vs = (si >= 0) ? *(const uint4*)(src_enc + (size_t)si * F + c) : z;
    uint4 vd = (di >= 0) ? *(const uint4*)(dst_enc + (size_t)di * F + c) : z;
    *(uint4*)(&A[e * 384 + c])       = vs;   // e_src  -> cols [0,128)
    *(uint4*)(&A[e * 384 + 256 + c]) = vd;   // e_dst  -> cols [256,384)
    float dx = 0.f, dy = 0.f;
    if (si >= 0) {
      dx = src_pos[si * 2 + 0] - dst_pos[di * 2 + 0];
      dy = src_pos[si * 2 + 1] - dst_pos[di * 2 + 1];
    }
#pragma unroll
    for (int j = 0; j < 8; ++j) {      // e_dist -> cols [128,256)
      float v = (si >= 0)
          ? dx * dist_w[c + j] + dy * dist_w[F + c + j] + dist_b[c + j] : 0.f;
      v = v > 0.f ? v : 0.f;
      A[e * 384 + F + c + j] = (_Float16)v;
    }
  }
  __syncthreads();
  int wave = tid >> 5, lane = tid & 31;
  int mlow = lane & 15;
  int mh   = (lane < 16) ? 0 : 8;
  int col  = wave * 16 + mlow;
  // GEMM1: [64,384] x [384,128]; 12 K-blocks x 4 M-tiles per wave
  {
    v8f acc[4] = {};
#pragma unroll
    for (int kb = 0; kb < 12; ++kb) {
      v16h b = *(const v16h*)(W1p + ((size_t)(wave * 12 + kb) * 32 + lane) * 16);
#pragma unroll
      for (int mt = 0; mt < 4; ++mt)
        acc[mt] = wmma16(load_a_frag(A + mt * 16 * 384, 384, kb, lane), b, acc[mt]);
    }
    float bv = b1[col];
#pragma unroll
    for (int mt = 0; mt < 4; ++mt) {
#pragma unroll
      for (int r = 0; r < 8; ++r) {
        float v = acc[mt][r] + bv;
        v = v > 0.f ? v : 0.f;
        H[(mt * 16 + r + mh) * F + col] = (_Float16)v;
      }
    }
  }
  __syncthreads();
  // GEMM2: [64,128] x [128,128] + scatter-add into aggregator
  {
    v8f acc[4] = {};
#pragma unroll
    for (int kb = 0; kb < 4; ++kb) {
      v16h b = *(const v16h*)(W2p + ((size_t)(wave * 4 + kb) * 32 + lane) * 16);
#pragma unroll
      for (int mt = 0; mt < 4; ++mt)
        acc[mt] = wmma16(load_a_frag(H + mt * 16 * F, F, kb, lane), b, acc[mt]);
    }
    float bv = b2[col];
#pragma unroll
    for (int mt = 0; mt < 4; ++mt) {
#pragma unroll
      for (int r = 0; r < 8; ++r) {
        int di = didx[mt * 16 + r + mh];
        if (di >= 0) atomicAdd(&agg[(size_t)di * F + col], acc[mt][r] + bv);
      }
    }
  }
}

// ---- post: LN + relu + @out_w + out_b + residual + relu; 64 rows/block -----
__global__ __launch_bounds__(256) void post_kernel(
    const float* __restrict__ agg,
    const float* __restrict__ ln_g, const float* __restrict__ ln_b,
    const _Float16* __restrict__ Wop, const float* __restrict__ out_b,
    const float* __restrict__ dst_feat, float* __restrict__ out) {
  __shared__ __align__(16) float    T[64 * F];     // 32 KB
  __shared__ __align__(16) _Float16 A[64 * F];     // 16 KB
  __shared__ float mu_s[64], rs_s[64];
  int row0 = blockIdx.x * 64;
  int tid  = threadIdx.x;
#pragma unroll
  for (int it = 0; it < 4; ++it) {
    int r = it * 16 + (tid >> 4), c0 = (tid & 15) * 8;
    const float* src = agg + (size_t)(row0 + r) * F + c0;
#pragma unroll
    for (int j = 0; j < 8; ++j) T[r * F + c0 + j] = src[j];
  }
  __syncthreads();
  if (tid < 64) {
    float s = 0.f;
    for (int c = 0; c < F; ++c) s += T[tid * F + c];
    float mu = s * (1.0f / F);
    float v = 0.f;
    for (int c = 0; c < F; ++c) { float d = T[tid * F + c] - mu; v += d * d; }
    mu_s[tid] = mu;
    rs_s[tid] = rsqrtf(v * (1.0f / F) + 1e-5f);
  }
  __syncthreads();
#pragma unroll
  for (int it = 0; it < 4; ++it) {
    int r = it * 16 + (tid >> 4), c0 = (tid & 15) * 8;
    float mu = mu_s[r], rs = rs_s[r];
#pragma unroll
    for (int j = 0; j < 8; ++j) {
      int c = c0 + j;
      float v = (T[r * F + c] - mu) * rs * ln_g[c] + ln_b[c];
      v = v > 0.f ? v : 0.f;
      A[r * F + c] = (_Float16)v;
    }
  }
  __syncthreads();
  int wave = tid >> 5, lane = tid & 31;
  int mlow = lane & 15;
  int mh   = (lane < 16) ? 0 : 8;
  v8f acc[4] = {};
#pragma unroll
  for (int kb = 0; kb < 4; ++kb) {
    v16h b = *(const v16h*)(Wop + ((size_t)(wave * 4 + kb) * 32 + lane) * 16);
#pragma unroll
    for (int mt = 0; mt < 4; ++mt)
      acc[mt] = wmma16(load_a_frag(A + mt * 16 * F, F, kb, lane), b, acc[mt]);
  }
  int col = wave * 16 + mlow;
  float bv = out_b[col];
#pragma unroll
  for (int mt = 0; mt < 4; ++mt) {
#pragma unroll
    for (int r = 0; r < 8; ++r) {
      size_t o = (size_t)(row0 + mt * 16 + r + mh) * F + col;
      float v = acc[mt][r] + bv + dst_feat[o];
      out[o] = v > 0.f ? v : 0.f;
    }
  }
}

// ---------------------------------------------------------------------------
extern "C" void kernel_launch(void* const* d_in, const int* in_sizes, int n_in,
                              void* d_out, int out_size, void* d_ws, size_t ws_size,
                              hipStream_t stream) {
  const float* src_feat = (const float*)d_in[0];
  const float* src_pos  = (const float*)d_in[1];
  const float* dst_feat = (const float*)d_in[2];
  const float* dst_pos  = (const float*)d_in[3];
  const float* src_w    = (const float*)d_in[4];
  const float* src_b    = (const float*)d_in[5];
  const float* dst_w    = (const float*)d_in[6];
  const float* dst_b    = (const float*)d_in[7];
  const float* dist_w   = (const float*)d_in[8];
  const float* dist_b   = (const float*)d_in[9];
  const float* edge_w1  = (const float*)d_in[10];
  const float* edge_b1  = (const float*)d_in[11];
  const float* edge_w2  = (const float*)d_in[12];
  const float* edge_b2  = (const float*)d_in[13];
  const float* ln_g     = (const float*)d_in[14];
  const float* ln_b     = (const float*)d_in[15];
  const float* out_w    = (const float*)d_in[16];
  const float* out_b    = (const float*)d_in[17];
  const int*   esrc     = (const int*)d_in[18];
  const int*   edst     = (const int*)d_in[19];
  const int E = in_sizes[18];
  const int S = in_sizes[0] / F;
  const int D = in_sizes[2] / F;

  // workspace carve-out (256B aligned): ~17 MB total
  char* ws = (char*)d_ws;
  size_t off = 0;
  auto carve = [&](size_t bytes) {
    void* p = ws + off;
    off = (off + bytes + 255) & ~(size_t)255;
    return p;
  };
  _Float16* src_enc = (_Float16*)carve((size_t)S * F * 2);
  _Float16* dst_enc = (_Float16*)carve((size_t)D * F * 2);
  float*    agg     = (float*)   carve((size_t)D * F * 4);
  _Float16* src_w_p = (_Float16*)carve((size_t)F * F * 2);
  _Float16* dst_w_p = (_Float16*)carve((size_t)F * F * 2);
  _Float16* w1_p    = (_Float16*)carve((size_t)3 * F * F * 2);
  _Float16* w2_p    = (_Float16*)carve((size_t)F * F * 2);
  _Float16* ow_p    = (_Float16*)carve((size_t)F * F * 2);

  pack_b<<<(F * F + 255) / 256, 256, 0, stream>>>(src_w,   src_w_p, F, F);
  pack_b<<<(F * F + 255) / 256, 256, 0, stream>>>(dst_w,   dst_w_p, F, F);
  pack_b<<<(3 * F * F + 255) / 256, 256, 0, stream>>>(edge_w1, w1_p, 3 * F, F);
  pack_b<<<(F * F + 255) / 256, 256, 0, stream>>>(edge_w2, w2_p, F, F);
  pack_b<<<(F * F + 255) / 256, 256, 0, stream>>>(out_w,   ow_p, F, F);

  node_encode<false><<<S / 64, 256, 0, stream>>>(src_feat, src_w_p, src_b,
                                                 src_enc, nullptr);
  node_encode<true><<<D / 64, 256, 0, stream>>>(dst_feat, dst_w_p, dst_b,
                                                dst_enc, agg);

  edge_mlp<<<(E + 63) / 64, 256, 0, stream>>>(src_enc, dst_enc, src_pos, dst_pos,
                                              dist_w, dist_b, w1_p, edge_b1,
                                              w2_p, edge_b2, esrc, edst, agg, E);

  post_kernel<<<D / 64, 256, 0, stream>>>(agg, ln_g, ln_b, ow_p, out_b,
                                          dst_feat, (float*)d_out);
}
